// ImportanceRouter_75617194213661
// MI455X (gfx1250) — matrix operand order repaired and checked by reading, not compiled, not run on hardware
//
#include <hip/hip_runtime.h>
#include <math.h>

typedef __attribute__((ext_vector_type(16))) _Float16 v16h;
typedef __attribute__((ext_vector_type(8)))  float    v8f;

#define NW 4  // waves per block in the MLP kernels

union F16Frag { float4 f[2]; v16h h; };

// Load a 32x16 f16 B fragment stored in per-lane-contiguous order (lane*16 elems).
__device__ __forceinline__ v16h load_bfrag(const _Float16* __restrict__ frag, int lane) {
  F16Frag u;
  const float4* p = reinterpret_cast<const float4*>(frag + lane * 16);
  u.f[0] = p[0];
  u.f[1] = p[1];
  return u.h;
}

// Load a 16x32 f16 A fragment from LDS: row base points at (row m, col c*32).
// Slots 0..7 hold K = h*8 + i, slots 8..15 hold K = 16 + h*8 + (i-8).
__device__ __forceinline__ v16h load_afrag(const _Float16* row_col_base, int h) {
  F16Frag u;
  u.f[0] = *reinterpret_cast<const float4*>(row_col_base + h * 8);
  u.f[1] = *reinterpret_cast<const float4*>(row_col_base + 16 + h * 8);
  return u.h;
}

__device__ __forceinline__ float gelu_exact(float x) {
  return 0.5f * x * (1.0f + erff(x * 0.70710678118654752f));
}
__device__ __forceinline__ float sigmoidf(float x) {
  return 1.0f / (1.0f + expf(-x));
}
__device__ __forceinline__ void wave_lds_sync() {
  asm volatile("s_wait_dscnt 0x0" ::: "memory");
}

// ---------------------------------------------------------------- utility ---
__global__ void zero_f32(float* p, int n) {
  int i = blockIdx.x * blockDim.x + threadIdx.x;
  if (i < n) p[i] = 0.0f;
}

__global__ void degree_accum(const int* __restrict__ ei, int n_edges, float* deg) {
  int i = blockIdx.x * blockDim.x + threadIdx.x;
  if (i < n_edges) {
    atomicAdd(&deg[ei[i]], 1.0f);
    atomicAdd(&deg[ei[n_edges + i]], 1.0f);
  }
}

__global__ void degree_max(const float* __restrict__ deg, int n, unsigned int* mx) {
  int i = blockIdx.x * blockDim.x + threadIdx.x;
  float m = 0.0f;
  for (int j = i; j < n; j += blockDim.x * gridDim.x) m = fmaxf(m, deg[j]);
  atomicMax(mx, __float_as_uint(m));  // deg >= 0 so bit order == float order
}

// Repack f32 weight matrix (row-major krows x 64) + bias into f16 WMMA B frags,
// padding K up to kc*32.  Row k < krows comes from W, row k == krows is the
// bias (driven by a constant-1.0 input feature), rows beyond are zero.
// Fragment f = c*4 + t covers K rows [c*32, c*32+32) and cols [t*16, t*16+16).
// Within a fragment: lane l (n = (l&15)+16t, h = l>>4) slot i -> K = c*32 + (i < 8 ? h*8 + i : 16 + h*8 + (i-8)).
__global__ void build_frags(const float* __restrict__ W, const float* __restrict__ bias,
                            _Float16* __restrict__ dst, int kc, int krows) {
  int tid = blockIdx.x * blockDim.x + threadIdx.x;
  int total = kc * 4 * 512;
  if (tid >= total) return;
  int f    = tid >> 9;
  int rem  = tid & 511;
  int lane = rem >> 4;
  int slot = rem & 15;
  int c = f >> 2, t = f & 3;
  int h = lane >> 4;
  int n = (lane & 15) + t * 16;
  int k = c * 32 + (slot < 8 ? h * 8 + slot : 16 + h * 8 + (slot - 8));
  float v = (k < krows) ? W[k * 64 + n] : ((k == krows) ? bias[n] : 0.0f);
  dst[tid] = (_Float16)v;
}

// ------------------------------------------------------------- node MLP -----
// Layer 1 input layout per row (K = 288): [256 features | deg tier 1.0 | 0...]
#define NTS 296  // node tile row stride (f16)
__global__ __launch_bounds__(32 * NW) void node_mlp_kernel(
    const float* __restrict__ nf, const float* __restrict__ deg,
    const unsigned int* __restrict__ maxbits, const int* __restrict__ tiers,
    const _Float16* __restrict__ w1f, const _Float16* __restrict__ w2f,
    const float* __restrict__ b2, const float* __restrict__ W3,
    const float* __restrict__ b3, float* __restrict__ out, int n_nodes) {
  __shared__ __align__(16) _Float16 s_tile[NW][16 * NTS];
  __shared__ __align__(16) _Float16 s_act1[NW][16 * 72];
  __shared__ __align__(16) float    s_act2[NW][16 * 68];

  const int wave = threadIdx.x >> 5;
  const int lane = threadIdx.x & 31;
  const int tile = blockIdx.x * NW + wave;
  if (tile * 16 >= n_nodes) return;
  const int row0 = tile * 16;
  const int h = lane >> 4;
  const int l15 = lane & 15;

  _Float16* T = s_tile[wave];
  // Stage 16 rows x 256 f32 -> f16 LDS (coalesced float4 loads)
  for (int it = lane; it < 16 * 64; it += 32) {
    int r = it >> 6, c4 = it & 63;
    float4 v = reinterpret_cast<const float4*>(nf + (size_t)(row0 + r) * 256)[c4];
    _Float16* d = T + r * NTS + c4 * 4;
    d[0] = (_Float16)v.x; d[1] = (_Float16)v.y;
    d[2] = (_Float16)v.z; d[3] = (_Float16)v.w;
  }
  // Zero the padded K region (cols 256..287), then drop in the 3 live extras.
  for (int it = lane; it < 16 * 4; it += 32) {
    int r = it >> 2, q = it & 3;
    *reinterpret_cast<float4*>(T + r * NTS + 256 + q * 8) = make_float4(0.f, 0.f, 0.f, 0.f);
  }
  if (lane < 16) {
    int nd = row0 + lane;
    float mx = __uint_as_float(*maxbits);
    _Float16* d = T + lane * NTS + 256;
    d[0] = (_Float16)(deg[nd] / (mx + 1e-10f));
    d[1] = (_Float16)(0.5f * (float)tiers[nd]);
    d[2] = (_Float16)1.0f;
  }
  wave_lds_sync();

  // Layer 1: 288 -> 64, bias folded into K via the constant-1 feature.
  v8f acc[4];
#pragma unroll
  for (int t = 0; t < 4; ++t)
#pragma unroll
    for (int v = 0; v < 8; ++v) acc[t][v] = 0.0f;
#pragma unroll
  for (int c = 0; c < 9; ++c) {
    v16h a = load_afrag(T + l15 * NTS + c * 32, h);
#pragma unroll
    for (int t = 0; t < 4; ++t) {
      v16h b = load_bfrag(w1f + (c * 4 + t) * 512, lane);
      acc[t] = __builtin_amdgcn_wmma_f32_16x16x32_f16(false, a, false, b,
                                                      (short)0, acc[t], false, false);
    }
  }
#pragma unroll
  for (int t = 0; t < 4; ++t) {
    int n = l15 + 16 * t;
#pragma unroll
    for (int v = 0; v < 8; ++v)
      s_act1[wave][(h * 8 + v) * 72 + n] = (_Float16)gelu_exact(acc[t][v]);
  }
  wave_lds_sync();

  // Layer 2: 64 -> 64
  v8f acc2[4];
#pragma unroll
  for (int t = 0; t < 4; ++t) {
    float bb = b2[l15 + 16 * t];
#pragma unroll
    for (int v = 0; v < 8; ++v) acc2[t][v] = bb;
  }
#pragma unroll
  for (int c = 0; c < 2; ++c) {
    v16h a = load_afrag(s_act1[wave] + l15 * 72 + c * 32, h);
#pragma unroll
    for (int t = 0; t < 4; ++t) {
      v16h b = load_bfrag(w2f + (c * 4 + t) * 512, lane);
      acc2[t] = __builtin_amdgcn_wmma_f32_16x16x32_f16(false, a, false, b,
                                                       (short)0, acc2[t], false, false);
    }
  }
#pragma unroll
  for (int t = 0; t < 4; ++t) {
    int n = l15 + 16 * t;
#pragma unroll
    for (int v = 0; v < 8; ++v)
      s_act2[wave][(h * 8 + v) * 68 + n] = gelu_exact(acc2[t][v]);
  }
  wave_lds_sync();

  // Layer 3: 64 -> 1, all 32 lanes (each half-sample), cross-half shuffle add.
  {
    const float* hr = s_act2[wave] + l15 * 68 + h * 32;
    float s = 0.0f;
#pragma unroll
    for (int j = 0; j < 32; ++j) s += hr[j] * W3[h * 32 + j];
    s += __shfl_xor(s, 16, 32);
    if (lane < 16) out[row0 + l15] = sigmoidf(s + b3[0]);
  }
}

// ------------------------------------------------------------- edge MLP -----
// Layer 1 input layout per row (K = 160): [128 features | src tgt 1.0 | 0...]
#define ETS 168  // edge tile row stride (f16)
__global__ __launch_bounds__(32 * NW) void edge_mlp_kernel(
    const float* __restrict__ ef, const int* __restrict__ eidx,
    const float* __restrict__ scores,
    const _Float16* __restrict__ w1f, const _Float16* __restrict__ w2f,
    const float* __restrict__ b2, const float* __restrict__ W3,
    const float* __restrict__ b3, float* __restrict__ out, int n_edges) {
  __shared__ __align__(16) _Float16 s_tile[NW][16 * ETS];
  __shared__ __align__(16) _Float16 s_act1[NW][16 * 72];
  __shared__ __align__(16) float    s_act2[NW][16 * 68];

  const int wave = threadIdx.x >> 5;
  const int lane = threadIdx.x & 31;
  const int tile = blockIdx.x * NW + wave;
  if (tile * 16 >= n_edges) return;
  const int row0 = tile * 16;
  const int h = lane >> 4;
  const int l15 = lane & 15;

  _Float16* T = s_tile[wave];
  // Stage 16 rows x 128 f32 -> f16 LDS
  for (int it = lane; it < 16 * 32; it += 32) {
    int r = it >> 5, c4 = it & 31;
    float4 v = reinterpret_cast<const float4*>(ef + (size_t)(row0 + r) * 128)[c4];
    _Float16* d = T + r * ETS + c4 * 4;
    d[0] = (_Float16)v.x; d[1] = (_Float16)v.y;
    d[2] = (_Float16)v.z; d[3] = (_Float16)v.w;
  }
  // Zero padded K region (cols 128..159), then the 3 live extras.
  for (int it = lane; it < 16 * 4; it += 32) {
    int r = it >> 2, q = it & 3;
    *reinterpret_cast<float4*>(T + r * ETS + 128 + q * 8) = make_float4(0.f, 0.f, 0.f, 0.f);
  }
  if (lane < 16) {
    int e = row0 + lane;
    _Float16* d = T + lane * ETS + 128;
    d[0] = (_Float16)scores[eidx[e]];            // src importance
    d[1] = (_Float16)scores[eidx[n_edges + e]];  // tgt importance
    d[2] = (_Float16)1.0f;
  }
  wave_lds_sync();

  // Layer 1: 160 -> 64, bias folded into K.
  v8f acc[4];
#pragma unroll
  for (int t = 0; t < 4; ++t)
#pragma unroll
    for (int v = 0; v < 8; ++v) acc[t][v] = 0.0f;
#pragma unroll
  for (int c = 0; c < 5; ++c) {
    v16h a = load_afrag(T + l15 * ETS + c * 32, h);
#pragma unroll
    for (int t = 0; t < 4; ++t) {
      v16h b = load_bfrag(w1f + (c * 4 + t) * 512, lane);
      acc[t] = __builtin_amdgcn_wmma_f32_16x16x32_f16(false, a, false, b,
                                                      (short)0, acc[t], false, false);
    }
  }
#pragma unroll
  for (int t = 0; t < 4; ++t) {
    int n = l15 + 16 * t;
#pragma unroll
    for (int v = 0; v < 8; ++v)
      s_act1[wave][(h * 8 + v) * 72 + n] = (_Float16)gelu_exact(acc[t][v]);
  }
  wave_lds_sync();

  // Layer 2: 64 -> 64
  v8f acc2[4];
#pragma unroll
  for (int t = 0; t < 4; ++t) {
    float bb = b2[l15 + 16 * t];
#pragma unroll
    for (int v = 0; v < 8; ++v) acc2[t][v] = bb;
  }
#pragma unroll
  for (int c = 0; c < 2; ++c) {
    v16h a = load_afrag(s_act1[wave] + l15 * 72 + c * 32, h);
#pragma unroll
    for (int t = 0; t < 4; ++t) {
      v16h b = load_bfrag(w2f + (c * 4 + t) * 512, lane);
      acc2[t] = __builtin_amdgcn_wmma_f32_16x16x32_f16(false, a, false, b,
                                                       (short)0, acc2[t], false, false);
    }
  }
#pragma unroll
  for (int t = 0; t < 4; ++t) {
    int n = l15 + 16 * t;
#pragma unroll
    for (int v = 0; v < 8; ++v)
      s_act2[wave][(h * 8 + v) * 68 + n] = gelu_exact(acc2[t][v]);
  }
  wave_lds_sync();

  // Layer 3: 64 -> 1, all 32 lanes + cross-half shuffle.
  {
    const float* hr = s_act2[wave] + l15 * 68 + h * 32;
    float s = 0.0f;
#pragma unroll
    for (int j = 0; j < 32; ++j) s += hr[j] * W3[h * 32 + j];
    s += __shfl_xor(s, 16, 32);
    if (lane < 16) out[row0 + l15] = sigmoidf(s + b3[0]);
  }
}

// ---------------------------------------------------------------- launch ----
extern "C" void kernel_launch(void* const* d_in, const int* in_sizes, int n_in,
                              void* d_out, int out_size, void* d_ws, size_t ws_size,
                              hipStream_t stream) {
  const float* nf    = (const float*)d_in[0];
  const float* ef    = (const float*)d_in[1];
  const int*   eidx  = (const int*)d_in[2];
  const int*   tiers = (const int*)d_in[3];
  const float* W1n = (const float*)d_in[4];
  const float* b1n = (const float*)d_in[5];
  const float* W2n = (const float*)d_in[6];
  const float* b2n = (const float*)d_in[7];
  const float* W3n = (const float*)d_in[8];
  const float* b3n = (const float*)d_in[9];
  const float* W1e = (const float*)d_in[10];
  const float* b1e = (const float*)d_in[11];
  const float* W2e = (const float*)d_in[12];
  const float* b2e = (const float*)d_in[13];
  const float* W3e = (const float*)d_in[14];
  const float* b3e = (const float*)d_in[15];

  const int n_nodes = in_sizes[0] / 256;
  const int n_edges = in_sizes[1] / 128;
  float* out = (float*)d_out;

  // Workspace carve-out (256-byte aligned chunks; frag buffers need 16B align)
  char* ws = (char*)d_ws;
  size_t off = 0;
  auto take = [&](size_t bytes) -> char* {
    char* p = ws + off;
    off = (off + bytes + 255) & ~(size_t)255;
    return p;
  };
  float*        deg  = (float*)take((size_t)n_nodes * sizeof(float));
  unsigned int* mx   = (unsigned int*)take(sizeof(unsigned int));
  _Float16*     w1nf = (_Float16*)take((size_t)9 * 4 * 512 * 2);  // K=288 (258+bias+pad)
  _Float16*     w2nf = (_Float16*)take((size_t)2 * 4 * 512 * 2);  // 64x64
  _Float16*     w1ef = (_Float16*)take((size_t)5 * 4 * 512 * 2);  // K=160 (130+bias+pad)
  _Float16*     w2ef = (_Float16*)take((size_t)2 * 4 * 512 * 2);  // 64x64

  // 1) degree = scatter-add of edge endpoints, then max
  zero_f32<<<(n_nodes + 255) / 256, 256, 0, stream>>>(deg, n_nodes);
  zero_f32<<<1, 32, 0, stream>>>((float*)mx, 1);
  degree_accum<<<(n_edges + 255) / 256, 256, 0, stream>>>(eidx, n_edges, deg);
  degree_max<<<256, 256, 0, stream>>>(deg, n_nodes, mx);

  // 2) repack weights (+bias row) into f16 WMMA B fragments
  build_frags<<<(9 * 4 * 512 + 255) / 256, 256, 0, stream>>>(W1n, b1n, w1nf, 9, 258);
  build_frags<<<(2 * 4 * 512 + 255) / 256, 256, 0, stream>>>(W2n, b2n, w2nf, 2, 64);
  build_frags<<<(5 * 4 * 512 + 255) / 256, 256, 0, stream>>>(W1e, b1e, w1ef, 5, 130);
  build_frags<<<(2 * 4 * 512 + 255) / 256, 256, 0, stream>>>(W2e, b2e, w2ef, 2, 64);

  // 3) node MLP -> out[0 : n_nodes)
  int ntiles = (n_nodes + 15) / 16;
  node_mlp_kernel<<<(ntiles + NW - 1) / NW, 32 * NW, 0, stream>>>(
      nf, deg, mx, tiers, w1nf, w2nf, b2n, W3n, b3n, out, n_nodes);

  // 4) edge MLP (gathers node scores from out) -> out[n_nodes : n_nodes+n_edges)
  int etiles = (n_edges + 15) / 16;
  edge_mlp_kernel<<<(etiles + NW - 1) / NW, 32 * NW, 0, stream>>>(
      ef, eidx, out, w1ef, w2ef, b2e, W3e, b3e, out + n_nodes, n_edges);
}
